// LSTMCell_68848325755666
// MI455X (gfx1250) — compile-verified, matrix operand order checked by reading, not run on hardware
//
#include <hip/hip_runtime.h>
#include <hip/hip_bf16.h>

// LSTM cell fused kernel for MI455X (gfx1250, wave32, WMMA).
// gates = x@Wx^T + h@Uh^T + b ; i,f,o = sigmoid; g = tanh;
// c = f*c_prev + i*g ; h = o*tanh(c)

typedef __attribute__((ext_vector_type(16))) __bf16 v16bf;
typedef __attribute__((ext_vector_type(8)))  __bf16 v8bf;
typedef __attribute__((ext_vector_type(8)))  float  v8f;

constexpr int BSZ = 4096;   // batch
constexpr int DD  = 1024;   // input dim
constexpr int HH  = 1024;   // hidden dim
constexpr int BM  = 128;    // block tile rows (batch)
constexpr int BN  = 64;     // block tile cols per gate (hidden)
constexpr int KK  = 32;     // K slice per step
constexpr int LDA = 40;     // LDS row stride in bf16 elems (80B: 16B-aligned, 20-bank stride)
constexpr int NSTEPS = (DD + HH) / KK;         // 64
constexpr int A_U32  = BM * LDA / 2;           // 2560 uints per A tile
constexpr int B_U32  = 4 * BN * LDA / 2;       // 5120 uints per B tile (4 gates)
constexpr int BUF_U32 = A_U32 + B_U32;         // 7680 uints per buffer

// CDNA5 packed f32->bf16 convert (assembles on gfx1250; no builtin declared).
static __device__ __forceinline__ unsigned pk_bf16(float lo, float hi) {
  unsigned r;
  asm("v_cvt_pk_bf16_f32 %0, %1, %2" : "=v"(r) : "v"(lo), "v"(hi));
  return r;
}

static __device__ __forceinline__ float fast_sigmoid(float x) {
  // 1/(1+exp(-x)) ; v_exp_f32 computes 2^x
  return __builtin_amdgcn_rcpf(1.0f + __builtin_amdgcn_exp2f(-1.44269504088896340736f * x));
}
static __device__ __forceinline__ float fast_tanh(float x) {
  // tanh(x) = 2*sigmoid(2x) - 1
  return 2.0f * __builtin_amdgcn_rcpf(1.0f + __builtin_amdgcn_exp2f(-2.88539008177792681472f * x)) - 1.0f;
}

__global__ __launch_bounds__(256)
void lstm_cell_wmma(const float* __restrict__ xp,  const float* __restrict__ hp,
                    const float* __restrict__ cp,
                    const float* __restrict__ Wi,  const float* __restrict__ bi,
                    const float* __restrict__ Wf,  const float* __restrict__ bf,
                    const float* __restrict__ Wg,  const float* __restrict__ bg,
                    const float* __restrict__ Wo,  const float* __restrict__ bo,
                    const float* __restrict__ Ui,  const float* __restrict__ ubi,
                    const float* __restrict__ Uf,  const float* __restrict__ ubf,
                    const float* __restrict__ Ug,  const float* __restrict__ ubg,
                    const float* __restrict__ Uo,  const float* __restrict__ ubo,
                    float* __restrict__ out) {
  __shared__ alignas(16) unsigned smem[2 * BUF_U32];   // 61440 B

  const int tid  = threadIdx.x;
  const int lane = tid & 31;
  const int wave = tid >> 5;          // 0..7
  const int wm   = wave & 3;          // 0..3 : 32-row slab
  const int wn   = wave >> 2;         // 0..1 : 32-col slab
  const int m0   = blockIdx.y * BM;   // batch tile origin
  const int n0   = blockIdx.x * BN;   // hidden tile origin

  // ---- global->register staging assignments ----
  // A: 2 threads per row, 16 floats each
  const int arow = tid >> 1;          // 0..127
  const int acg  = (tid & 1) * 16;    // col group {0,16}
  // B: every thread loads 8 floats from EACH of the 4 gate matrices
  //    (gate loop unrolled at compile time -> all bases stay scalar)
  const int brow = tid >> 2;          // 0..63 : row within gate tile
  const int bcg  = (tid & 3) * 8;     // col group {0,8,16,24}

  // loop-invariant row offsets (elements)
  const long aoff = (long)(m0 + arow) * DD + acg;
  const long boff = (long)(n0 + brow) * DD + bcg;

  float4 ar[4];      // 16 floats of A slice
  float4 br[4][2];   // 8 floats per gate

  auto loadStage = [&](int s) {
    const int k  = (s & 31) * KK;
    const bool p = (s >= 32);
    const float* Ap = p ? hp : xp;
    const float* Bp[4];
    Bp[0] = p ? Ui : Wi;
    Bp[1] = p ? Uf : Wf;
    Bp[2] = p ? Ug : Wg;
    Bp[3] = p ? Uo : Wo;
    const float* ap = Ap + aoff + k;
    #pragma unroll
    for (int i = 0; i < 4; ++i) ar[i] = *(const float4*)(ap + 4 * i);
    #pragma unroll
    for (int g = 0; g < 4; ++g) {
      const float* bp = Bp[g] + boff + k;
      br[g][0] = *(const float4*)(bp);
      br[g][1] = *(const float4*)(bp + 4);
    }
  };

  auto storeStage = [&](int buf) {
    unsigned* Au = smem + buf * BUF_U32;
    unsigned* Bu = Au + A_U32;
    unsigned ua[8];
    #pragma unroll
    for (int i = 0; i < 4; ++i) {
      ua[2 * i + 0] = pk_bf16(ar[i].x, ar[i].y);
      ua[2 * i + 1] = pk_bf16(ar[i].z, ar[i].w);
    }
    uint4* da = (uint4*)(Au + arow * (LDA / 2) + (acg >> 1));
    da[0] = make_uint4(ua[0], ua[1], ua[2], ua[3]);
    da[1] = make_uint4(ua[4], ua[5], ua[6], ua[7]);
    #pragma unroll
    for (int g = 0; g < 4; ++g) {
      unsigned u0 = pk_bf16(br[g][0].x, br[g][0].y);
      unsigned u1 = pk_bf16(br[g][0].z, br[g][0].w);
      unsigned u2 = pk_bf16(br[g][1].x, br[g][1].y);
      unsigned u3 = pk_bf16(br[g][1].z, br[g][1].w);
      *(uint4*)(Bu + g * (BN * LDA / 2) + brow * (LDA / 2) + (bcg >> 1)) =
          make_uint4(u0, u1, u2, u3);
    }
  };

  // ---- accumulators: 4 gates x 2 mi x 2 ni, each 16x16 f32 (8 VGPRs) ----
  v8f acc[4][2][2];
  #pragma unroll
  for (int g = 0; g < 4; ++g)
    #pragma unroll
    for (int mi = 0; mi < 2; ++mi)
      #pragma unroll
      for (int ni = 0; ni < 2; ++ni)
        acc[g][mi][ni] = (v8f)(0.0f);

  const int r  = lane & 15;
  const int hf = lane >> 4;

  loadStage(0);
  storeStage(0);
  __syncthreads();

  for (int s = 0; s < NSTEPS; ++s) {
    const int buf = s & 1;
    if (s + 1 < NSTEPS) loadStage(s + 1);          // hide global latency under WMMA

    const __bf16* Abf = (const __bf16*)(smem + buf * BUF_U32);
    const __bf16* Bbf = Abf + BM * LDA;

    // Preload ALL fragments (2 A + 8 B) so the 16 WMMAs can issue
    // back-to-back behind a single DS wait instead of pairwise stalls.
    v16bf afr[2];
    #pragma unroll
    for (int mi = 0; mi < 2; ++mi) {
      const int off = (wm * 32 + mi * 16 + r) * LDA + hf * 8;
      ((v8bf*)&afr[mi])[0] = *(const v8bf*)(Abf + off);
      ((v8bf*)&afr[mi])[1] = *(const v8bf*)(Abf + off + 16);
    }
    v16bf bfr[4][2];
    #pragma unroll
    for (int g = 0; g < 4; ++g)
      #pragma unroll
      for (int ni = 0; ni < 2; ++ni) {
        const int off = (g * BN + wn * 32 + ni * 16 + r) * LDA + hf * 8;
        ((v8bf*)&bfr[g][ni])[0] = *(const v8bf*)(Bbf + off);
        ((v8bf*)&bfr[g][ni])[1] = *(const v8bf*)(Bbf + off + 16);
      }

    #pragma unroll
    for (int g = 0; g < 4; ++g)
      #pragma unroll
      for (int ni = 0; ni < 2; ++ni)
        #pragma unroll
        for (int mi = 0; mi < 2; ++mi)
          acc[g][mi][ni] = __builtin_amdgcn_wmma_f32_16x16x32_bf16(
              false, afr[mi], false, bfr[g][ni], (short)0, acc[g][mi][ni], false, false);

    if (s + 1 < NSTEPS) storeStage(buf ^ 1);       // fill other buffer
    __syncthreads();
  }

  // ---- epilogue: biases, activations, cell update (fully local) ----
  const int m0g = m0 + wm * 32;
  const int n0g = n0 + wn * 32;
  float* out_h = out;
  float* out_c = out + (long)BSZ * HH;

  #pragma unroll
  for (int ni = 0; ni < 2; ++ni) {
    const int col = n0g + ni * 16 + (lane & 15);
    const float bsi = bi[col] + ubi[col];
    const float bsf = bf[col] + ubf[col];
    const float bsg = bg[col] + ubg[col];
    const float bso = bo[col] + ubo[col];
    #pragma unroll
    for (int mi = 0; mi < 2; ++mi) {
      const int rbase = m0g + mi * 16 + (lane >> 4) * 8;
      #pragma unroll
      for (int v = 0; v < 8; ++v) {
        const int  row = rbase + v;
        const long idx = (long)row * HH + col;
        const float it = fast_sigmoid(acc[0][mi][ni][v] + bsi);
        const float ft = fast_sigmoid(acc[1][mi][ni][v] + bsf);
        const float gt = fast_tanh   (acc[2][mi][ni][v] + bsg);
        const float ot = fast_sigmoid(acc[3][mi][ni][v] + bso);
        const float ct = ft * cp[idx] + it * gt;
        out_c[idx] = ct;
        out_h[idx] = ot * fast_tanh(ct);
      }
    }
  }
}

extern "C" void kernel_launch(void* const* d_in, const int* in_sizes, int n_in,
                              void* d_out, int out_size, void* d_ws, size_t ws_size,
                              hipStream_t stream) {
  (void)in_sizes; (void)n_in; (void)out_size; (void)d_ws; (void)ws_size;
  const float* xp  = (const float*)d_in[0];
  const float* hp  = (const float*)d_in[1];
  const float* cp  = (const float*)d_in[2];
  const float* Wi  = (const float*)d_in[3];  const float* bi  = (const float*)d_in[4];
  const float* Wf  = (const float*)d_in[5];  const float* bf  = (const float*)d_in[6];
  const float* Wg  = (const float*)d_in[7];  const float* bg  = (const float*)d_in[8];
  const float* Wo  = (const float*)d_in[9];  const float* bo  = (const float*)d_in[10];
  const float* Ui  = (const float*)d_in[11]; const float* ubi = (const float*)d_in[12];
  const float* Uf  = (const float*)d_in[13]; const float* ubf = (const float*)d_in[14];
  const float* Ug  = (const float*)d_in[15]; const float* ubg = (const float*)d_in[16];
  const float* Uo  = (const float*)d_in[17]; const float* ubo = (const float*)d_in[18];
  float* out = (float*)d_out;

  dim3 grid(HH / BN, BSZ / BM);   // (16, 32)
  dim3 block(256);
  lstm_cell_wmma<<<grid, block, 0, stream>>>(
      xp, hp, cp, Wi, bi, Wf, bf, Wg, bg, Wo, bo,
      Ui, ubi, Uf, ubf, Ug, ubg, Uo, ubo, out);
}
